// BrainSegNet_69398081569397
// MI455X (gfx1250) — matrix operand order repaired and checked by reading, not compiled, not run on hardware
//
#include <hip/hip_runtime.h>
#include <hip/hip_bf16.h>
#include <math.h>

typedef __attribute__((ext_vector_type(16))) _Float16 v16h;
typedef __attribute__((ext_vector_type(8)))  float    v8f;

static __device__ __forceinline__ int imin_d(int a, int b) { return a < b ? a : b; }

// ---------------------------------------------------------------------------
// FAST conv (requires Fi = power of two, Fi >= 32; K = 7*Fi, K % 32 == 0).
// Block = 128 threads (4 wave32). Tile: 32 rows (M) x 64 cols (Fo), K step 32.
// j = k0 >> lgFi is uniform per K-step (tile never straddles a neighbor slot):
// gather = one no[] lookup + two aligned float4 loads per thread.
// LDS double-buffered: ONE barrier per K-step; next gather issued before the
// current WMMAs so global latency overlaps the matrix math.
// ---------------------------------------------------------------------------
__global__ __launch_bounds__(128) void conv_wmma_fast_kernel(
    const float* __restrict__ X, const int* __restrict__ no,
    const float* __restrict__ W, const float* __restrict__ Bb,
    float* __restrict__ H, int Nrows, int lgFi, int K, int Fo)
{
    __shared__ _Float16 As[2][32][40];   // double buffer; row stride 80B (16B aligned)

    const int t    = threadIdx.x;
    const int wave = t >> 5;
    const int lane = t & 31;
    const int m0   = blockIdx.x * 32;
    const int n0   = blockIdx.y * 64;
    const int Fim1 = (1 << lgFi) - 1;

    // gather mapping: 32 rows x 32 k, 8 elems (2 float4) per thread
    const int grow = t >> 2;               // 0..31
    const int kq   = (t & 3) << 3;         // 0,8,16,24
    const int mld  = imin_d(m0 + grow, Nrows - 1);
    const int nb7  = mld * 7;

    // B fragment mapping
    const int ncol    = n0 + wave * 16 + (lane & 15);
    const float* wrow = W + (size_t)imin_d(ncol, Fo - 1) * K;  // clamped: safe loads,
                                                               // garbage cols never stored
    const int khi16 = (lane >> 4) << 4;   // 0/16: B K-offset per lane-half
    const int khi8  = (lane >> 4) << 3;   // 0/8 : unused; A also uses khi16
    const int arow  = lane & 15;

    v8f acc0 = {}, acc1 = {};

    // ---- prologue: gather tile 0 into buffer 0 ----
    {
        const int j    = 0;
        const float* src = X + (size_t)no[nb7 + j] * (Fim1 + 1) + kq;
        float4 p0 = *reinterpret_cast<const float4*>(src);
        float4 p1 = *reinterpret_cast<const float4*>(src + 4);
        As[0][grow][kq + 0] = (_Float16)p0.x;  As[0][grow][kq + 1] = (_Float16)p0.y;
        As[0][grow][kq + 2] = (_Float16)p0.z;  As[0][grow][kq + 3] = (_Float16)p0.w;
        As[0][grow][kq + 4] = (_Float16)p1.x;  As[0][grow][kq + 5] = (_Float16)p1.y;
        As[0][grow][kq + 6] = (_Float16)p1.z;  As[0][grow][kq + 7] = (_Float16)p1.w;
    }
    __syncthreads();

    int buf = 0;
    for (int k0 = 0; k0 < K; k0 += 32, buf ^= 1) {
        const bool more = (k0 + 32) < K;

        // ---- issue next tile's gather loads early (overlap with WMMA) ----
        float4 q0, q1;
        if (more) {
            const int kn   = k0 + 32;
            const int j    = kn >> lgFi;            // uniform
            const int foff = kn & ~Fim1 & (Fim1);   // always 0 (unused)
            const float* src = X + (((size_t)no[nb7 + j]) << lgFi) + (kn & Fim1) + kq;
            (void)foff;
            q0 = *reinterpret_cast<const float4*>(src);
            q1 = *reinterpret_cast<const float4*>(src + 4);
        }

        // ---- B fragment: 16 consecutive k, 4 aligned float4 loads ----
        v16h b;
        {
            const float4* wv = reinterpret_cast<const float4*>(wrow + k0 + khi16);
            float4 w0 = wv[0], w1 = wv[1], w2 = wv[2], w3 = wv[3];
            b[0]=(_Float16)w0.x; b[1]=(_Float16)w0.y; b[2]=(_Float16)w0.z; b[3]=(_Float16)w0.w;
            b[4]=(_Float16)w1.x; b[5]=(_Float16)w1.y; b[6]=(_Float16)w1.z; b[7]=(_Float16)w1.w;
            b[8]=(_Float16)w2.x; b[9]=(_Float16)w2.y; b[10]=(_Float16)w2.z; b[11]=(_Float16)w2.w;
            b[12]=(_Float16)w3.x; b[13]=(_Float16)w3.y; b[14]=(_Float16)w3.z; b[15]=(_Float16)w3.w;
        }

        // ---- A fragments from current buffer ----
        v16h a0, a1;
        #pragma unroll
        for (int i = 0; i < 16; ++i) {
            int kl = khi16 + i;   // contiguous k = 16h + i, matches B
            a0[i] = As[buf][arow][kl];
            a1[i] = As[buf][16 + arow][kl];
        }

        acc0 = __builtin_amdgcn_wmma_f32_16x16x32_f16(false, a0, false, b, (short)0, acc0, false, false);
        acc1 = __builtin_amdgcn_wmma_f32_16x16x32_f16(false, a1, false, b, (short)0, acc1, false, false);

        // ---- commit prefetched tile into the other buffer ----
        if (more) {
            const int nb = buf ^ 1;
            As[nb][grow][kq + 0] = (_Float16)q0.x;  As[nb][grow][kq + 1] = (_Float16)q0.y;
            As[nb][grow][kq + 2] = (_Float16)q0.z;  As[nb][grow][kq + 3] = (_Float16)q0.w;
            As[nb][grow][kq + 4] = (_Float16)q1.x;  As[nb][grow][kq + 5] = (_Float16)q1.y;
            As[nb][grow][kq + 6] = (_Float16)q1.z;  As[nb][grow][kq + 7] = (_Float16)q1.w;
        }
        __syncthreads();   // separates this step's As[buf] reads from next step's writes
    }

    if (ncol < Fo) {
        const float bias  = Bb[ncol];
        const int   rbase = m0 + ((lane >> 4) << 3);
        #pragma unroll
        for (int r = 0; r < 8; ++r) {
            int orow = rbase + r;
            if (orow < Nrows)        H[(size_t)orow * Fo + ncol]        = acc0[r] + bias;
            if (orow + 16 < Nrows)   H[(size_t)(orow + 16) * Fo + ncol] = acc1[r] + bias;
        }
    }
}

// ---------------------------------------------------------------------------
// GENERIC conv (any Fi/K; used only for layer 0: Fi=3, K=21 -> one K-step).
// ---------------------------------------------------------------------------
__global__ __launch_bounds__(128) void conv_wmma_generic_kernel(
    const float* __restrict__ X, const int* __restrict__ no,
    const float* __restrict__ W, const float* __restrict__ Bb,
    float* __restrict__ H, int Nrows, int Fi, int K, int Fo)
{
    __shared__ _Float16 As[16][40];

    const int t    = threadIdx.x;
    const int wave = t >> 5;
    const int lane = t & 31;
    const int m0   = blockIdx.x * 16;
    const int n0   = blockIdx.y * 64;

    const int arow_ld = t >> 3;
    const int kq      = (t & 7) << 2;
    const int mld     = imin_d(m0 + arow_ld, Nrows - 1);
    const int nb7     = mld * 7;

    const int ncol  = n0 + wave * 16 + (lane & 15);
    const bool nok  = (ncol < Fo);
    const float* wrow = W + (size_t)imin_d(ncol, Fo - 1) * K;
    const int khi16 = (lane >> 4) << 4;
    const int khi8  = (lane >> 4) << 3;
    const int arow  = lane & 15;

    v8f acc = {};

    for (int k0 = 0; k0 < K; k0 += 32) {
        #pragma unroll
        for (int u = 0; u < 4; ++u) {
            int k = k0 + kq + u;
            float v = 0.0f;
            if (k < K) {
                int j = k / Fi;
                int f = k - j * Fi;
                v = X[(size_t)no[nb7 + j] * Fi + f];
            }
            As[arow_ld][kq + u] = (_Float16)v;
        }
        __syncthreads();

        v16h a, b;
        #pragma unroll
        for (int i = 0; i < 16; ++i) {
            int kl = khi16 + i;
            a[i] = As[arow][kl];
            int kg = k0 + khi16 + i;
            float wv = (nok && kg < K) ? wrow[kg] : 0.0f;
            b[i] = (_Float16)wv;
        }
        acc = __builtin_amdgcn_wmma_f32_16x16x32_f16(false, a, false, b, (short)0, acc, false, false);
        __syncthreads();
    }

    if (ncol < Fo) {
        const float bias  = Bb[ncol];
        const int   rbase = m0 + ((lane >> 4) << 3);
        #pragma unroll
        for (int r = 0; r < 8; ++r) {
            int orow = rbase + r;
            if (orow < Nrows)
                H[(size_t)orow * Fo + ncol] = acc[r] + bias;
        }
    }
}

// ---------------------------------------------------------------------------
// BN stats: one block per channel; population mean/var over rows.
// ---------------------------------------------------------------------------
__global__ __launch_bounds__(256) void bn_stats_kernel(
    const float* __restrict__ H, int Nrows, int Fo,
    const float* __restrict__ g, const float* __restrict__ bb,
    float* __restrict__ scale, float* __restrict__ shift)
{
    __shared__ float ssum[256], ssq[256];
    const int c = blockIdx.x;
    float s = 0.0f, q = 0.0f;
    for (int r = threadIdx.x; r < Nrows; r += 256) {
        float v = H[(size_t)r * Fo + c];
        s += v; q += v * v;
    }
    ssum[threadIdx.x] = s; ssq[threadIdx.x] = q;
    __syncthreads();
    for (int off = 128; off > 0; off >>= 1) {
        if (threadIdx.x < off) {
            ssum[threadIdx.x] += ssum[threadIdx.x + off];
            ssq [threadIdx.x] += ssq [threadIdx.x + off];
        }
        __syncthreads();
    }
    if (threadIdx.x == 0) {
        float inv  = 1.0f / (float)Nrows;
        float mean = ssum[0] * inv;
        float var  = ssq[0] * inv - mean * mean;
        float sc   = g[c] * rsqrtf(var + 1e-5f);
        scale[c] = sc;
        shift[c] = bb[c] - mean * sc;
    }
}

__global__ void bn_apply_relu_kernel(const float* __restrict__ H,
                                     const float* __restrict__ scale,
                                     const float* __restrict__ shift,
                                     float* __restrict__ Y, int total, int Fo)
{
    int e = blockIdx.x * blockDim.x + threadIdx.x;
    if (e >= total) return;
    int c = e % Fo;
    float v = H[e] * scale[c] + shift[c];
    Y[e] = v > 0.0f ? v : 0.0f;
}

// ---------------------------------------------------------------------------
// Pool: reference's reshape-based max (flat element i*7F + c*7 + j); argmax
// keeps FIRST max.
// ---------------------------------------------------------------------------
__global__ void pool_kernel(const float* __restrict__ X, const int* __restrict__ no,
                            float* __restrict__ P, int* __restrict__ IDX,
                            int n_out, int F)
{
    int e = blockIdx.x * blockDim.x + threadIdx.x;
    if (e >= n_out * F) return;
    int i = e / F, c = e - i * F;
    long base = (long)i * 7 * F + (long)c * 7;
    float best = -INFINITY; int bj = 0;
    #pragma unroll
    for (int j = 0; j < 7; ++j) {
        long tt = base + j;
        int r = (int)(tt / F);
        int f = (int)(tt - (long)r * F);
        float v = X[(size_t)no[r] * F + f];
        if (v > best) { best = v; bj = j; }
    }
    P[e] = best;
    IDX[e] = bj;
}

__global__ void fzero_kernel(float* __restrict__ p, int n)
{
    int e = blockIdx.x * blockDim.x + threadIdx.x;
    if (e < n) p[e] = 0.0f;
}

// ---------------------------------------------------------------------------
// Unpool scatter: col = no[r*7 + idx[e]], row = floor(e * (feat/(raw*feat-1)))
// (numpy linspace semantics: step computed first in double; last -> feat-1).
// ---------------------------------------------------------------------------
__global__ void unpool_scatter_kernel(const float* __restrict__ Xs,
                                      const int* __restrict__ IDX,
                                      const int* __restrict__ no,
                                      float* __restrict__ OUT,
                                      int raw, int feat)
{
    long total = (long)raw * feat;
    long e = (long)blockIdx.x * blockDim.x + threadIdx.x;
    if (e >= total) return;
    int r   = (int)(e / feat);
    int col = no[r * 7 + IDX[e]];
    int rowi;
    if (e == total - 1) rowi = feat - 1;
    else {
        double step = (double)feat / (double)(total - 1);
        rowi = (int)((double)e * step);
        if (rowi > feat - 1) rowi = feat - 1;
    }
    OUT[(size_t)col * feat + rowi] = Xs[e];
}

// ---------------------------------------------------------------------------
// Host orchestration
// ---------------------------------------------------------------------------
struct Spec { int fi, fo, bn; };
static const Spec SPECS[21] = {
    {3,64,1},{64,64,1},{64,128,1},{128,128,1},{128,256,1},{256,256,1},
    {256,512,1},{512,512,1},{512,512,1},{512,512,1},{512,512,1},{512,512,1},
    {512,512,1},{512,256,1},{256,256,1},{256,128,1},{128,128,1},{128,64,1},
    {64,64,1},{64,64,1},{64,36,0}
};

extern "C" void kernel_launch(void* const* d_in, const int* in_sizes, int n_in,
                              void* d_out, int out_size, void* d_ws, size_t ws_size,
                              hipStream_t stream)
{
    (void)in_sizes; (void)out_size; (void)ws_size;

    const float* x_in = (const float*)d_in[0];
    const float *Wp[21], *bp[21], *gp[21], *bbp[21];
    int pi = 1;
    for (int l = 0; l < 21; ++l) {
        Wp[l] = (const float*)d_in[pi++];
        bp[l] = (const float*)d_in[pi++];
        if (SPECS[l].bn) { gp[l] = (const float*)d_in[pi++]; bbp[l] = (const float*)d_in[pi++]; }
        else             { gp[l] = nullptr; bbp[l] = nullptr; }
    }
    const int* no10242 = (const int*)d_in[n_in - 4];
    const int* no2562  = (const int*)d_in[n_in - 3];
    const int* no642   = (const int*)d_in[n_in - 2];
    const int* no162   = (const int*)d_in[n_in - 1];

    const size_t BUFE = (size_t)10242 * 512;
    float* bufA  = (float*)d_ws;
    float* bufB  = bufA + BUFE;
    float* scale = bufB + BUFE;
    float* shift = scale + 512;
    int*   i1 = (int*)(shift + 512);
    int*   i2 = i1 + 2562 * 64;
    int*   i3 = i2 + 642 * 128;
    int*   i4 = i3 + 162 * 512;

    auto conv = [&](const float* X, const int* no, int l, int Nr, float* H) {
        const Spec& s = SPECS[l];
        int K = 7 * s.fi;
        if (s.fi >= 32 && (s.fi & (s.fi - 1)) == 0) {
            int lg = __builtin_ctz((unsigned)s.fi);
            dim3 grid((Nr + 31) / 32, (s.fo + 63) / 64);
            conv_wmma_fast_kernel<<<grid, 128, 0, stream>>>(X, no, Wp[l], bp[l], H, Nr, lg, K, s.fo);
        } else {
            dim3 grid((Nr + 15) / 16, (s.fo + 63) / 64);
            conv_wmma_generic_kernel<<<grid, 128, 0, stream>>>(X, no, Wp[l], bp[l], H, Nr, s.fi, K, s.fo);
        }
    };
    auto bn = [&](float* H, int l, int Nr) {
        const Spec& s = SPECS[l];
        bn_stats_kernel<<<s.fo, 256, 0, stream>>>(H, Nr, s.fo, gp[l], bbp[l], scale, shift);
        int tot = Nr * s.fo;
        bn_apply_relu_kernel<<<(tot + 255) / 256, 256, 0, stream>>>(H, scale, shift, H, tot, s.fo);
    };
    auto pool = [&](const float* X, const int* no, int n_out, int F, float* P, int* IDX) {
        int tot = n_out * F;
        pool_kernel<<<(tot + 255) / 256, 256, 0, stream>>>(X, no, P, IDX, n_out, F);
    };
    auto unpool = [&](const float* Xs, const int* IDX, const int* no,
                      int raw, int feat, int nbig, float* OUT) {
        int ztot = nbig * feat;
        fzero_kernel<<<(ztot + 255) / 256, 256, 0, stream>>>(OUT, ztot);
        int tot = raw * feat;
        unpool_scatter_kernel<<<(tot + 255) / 256, 256, 0, stream>>>(Xs, IDX, no, OUT, raw, feat);
    };

    // ---- forward ----
    conv(x_in, no10242, 0, 10242, bufA);  bn(bufA, 0, 10242);      // c1_1
    conv(bufA, no10242, 1, 10242, bufB);  bn(bufB, 1, 10242);      // c1_2
    pool(bufB, no10242, 2562, 64, bufA, i1);

    conv(bufA, no2562, 2, 2562, bufB);    bn(bufB, 2, 2562);       // c2_1
    conv(bufB, no2562, 3, 2562, bufA);    bn(bufA, 3, 2562);       // c2_2
    pool(bufA, no2562, 642, 128, bufB, i2);

    conv(bufB, no642, 4, 642, bufA);      bn(bufA, 4, 642);        // c3_1
    conv(bufA, no642, 5, 642, bufB);      bn(bufB, 5, 642);        // c3_2
    conv(bufB, no642, 6, 642, bufA);      bn(bufA, 6, 642);        // c3_3
    pool(bufA, no642, 162, 512, bufB, i3);

    conv(bufB, no162, 7, 162, bufA);      bn(bufA, 7, 162);        // c4_1
    conv(bufA, no162, 8, 162, bufB);      bn(bufB, 8, 162);        // c4_2
    conv(bufB, no162, 9, 162, bufA);      bn(bufA, 9, 162);        // c4_3
    pool(bufA, no162, 42, 512, bufB, i4);

    unpool(bufB, i4, no162, 42, 512, 162, bufA);
    conv(bufA, no162, 10, 162, bufB);     bn(bufB, 10, 162);       // c6_1
    conv(bufB, no162, 11, 162, bufA);     bn(bufA, 11, 162);       // c6_2
    conv(bufA, no162, 12, 162, bufB);     bn(bufB, 12, 162);       // c6_3

    unpool(bufB, i3, no642, 162, 512, 642, bufA);
    conv(bufA, no642, 13, 642, bufB);     bn(bufB, 13, 642);       // c7_1
    conv(bufB, no642, 14, 642, bufA);     bn(bufA, 14, 642);       // c7_2
    conv(bufA, no642, 15, 642, bufB);     bn(bufB, 15, 642);       // c7_3

    unpool(bufB, i2, no2562, 642, 128, 2562, bufA);
    conv(bufA, no2562, 16, 2562, bufB);   bn(bufB, 16, 2562);      // c8_1
    conv(bufB, no2562, 17, 2562, bufA);   bn(bufA, 17, 2562);      // c8_2

    unpool(bufA, i1, no10242, 2562, 64, 10242, bufB);
    conv(bufB, no10242, 18, 10242, bufA); bn(bufA, 18, 10242);     // c9_1
    conv(bufA, no10242, 19, 10242, bufB); bn(bufB, 19, 10242);     // c9_2

    conv(bufB, no10242, 20, 10242, (float*)d_out);                 // c10 (no BN)
}
